// Block_QKA_2628519985650
// MI455X (gfx1250) — matrix operand order repaired and checked by reading, not compiled, not run on hardware
//
#include <hip/hip_runtime.h>
#include <hip/hip_bf16.h>

typedef __attribute__((ext_vector_type(16))) _Float16 v16h;
typedef __attribute__((ext_vector_type(8)))  float    v8f;

// ---------------------------------------------------------------------------
// Problem constants (reference: T=4, B=16, C=384, H=W=16 -> N=256, hid=1536)
// ---------------------------------------------------------------------------
#define TT   4
#define BB   16
#define CC   384
#define NN   256
#define HID  1536
#define NB16 16          // N / 16 n-tiles
#define BN_EPS 1e-5f
#define MW   2           // m-tiles per wave
#define NW   2           // n-tiles per wave

// ---------------------------------------------------------------------------
// Blocked spike layout ("B-operand layout" for v_wmma_f32_16x16x32_f16):
// tile (kb = k/32, nb = n/16) of 512 f16, per-lane 16 contiguous f16:
//   lane = ((k>>4)&1)*16 + (n&15), element i = k&15  (K = khalf*16 + i)
// ---------------------------------------------------------------------------
__device__ __forceinline__ size_t sblk(int k, int n) {
    return (size_t)(((k >> 5) * NB16 + (n >> 4)) * 512
                    + (((k >> 4) & 1) * 256) + ((n & 15) * 16) + (k & 15));
}

// ---------------------------------------------------------------------------
// Weight pack: f32 row-major (M x K) -> f16 A-operand blocked layout.
// tile (mtile, kb): per-lane 16 f16; lane m = lane&15, khalf = lane>>4,
// element i -> K = kb*32 + khalf*16 + i
// ---------------------------------------------------------------------------
__global__ void pack_w(const float* __restrict__ W, _Float16* __restrict__ out,
                       int M, int K) {
    int tid = blockIdx.x * blockDim.x + threadIdx.x;
    if (tid >= M * K) return;
    int i    = tid & 15;
    int lane = (tid >> 4) & 31;
    int tIdx = tid >> 9;
    int KB   = K >> 5;
    int mtile = tIdx / KB;
    int kb    = tIdx % KB;
    int m = mtile * 16 + (lane & 15);
    int k = kb * 32 + ((lane >> 4) * 16) + i;
    out[tid] = (_Float16)W[(size_t)m * K + k];
}

// ---------------------------------------------------------------------------
// LIF on the raw input x (f32, (T,B,C,N) row-major) -> spikes into blocked f16
// tau=2 -> v = 0.5*(v + x); spike = (v >= 1); hard reset.
// ---------------------------------------------------------------------------
__global__ void lif_input(const float* __restrict__ x, _Float16* __restrict__ xs) {
    int tid = blockIdx.x * blockDim.x + threadIdx.x;
    if (tid >= BB * CC * NN) return;
    int n = tid & (NN - 1);
    int c = (tid >> 8) % CC;
    int b = tid / (CC * NN);
    size_t so = sblk(c, n);
    float v = 0.f;
#pragma unroll
    for (int t = 0; t < TT; ++t) {
        float xv = x[(((size_t)t * BB + b) * CC + c) * NN + n];
        v = 0.5f * (v + xv);
        float s = (v >= 1.0f) ? 1.0f : 0.0f;
        v *= (1.0f - s);
        xs[(size_t)(t * BB + b) * CC * NN + so] = (_Float16)s;
    }
}

// ---------------------------------------------------------------------------
// Fused WMMA GEMM + BN (+bias) [+residual] [+LIF over T in registers].
// One wave owns a (MW*16 x NW*16) output block for all 4 time steps:
// 16 v8f accumulators; LIF scan runs entirely in VGPRs in the epilogue.
// Wb: f16 A-blocked (M x K).  Sb: f16 B-blocked spikes (T,B,K,N).
// outSpk: f16 B-blocked (T,B,M,N).  outF32/resid: f32 row-major (T,B,M,N).
// ---------------------------------------------------------------------------
__global__ __launch_bounds__(32)
void gemm_bn_lif(const _Float16* __restrict__ Wb, const _Float16* __restrict__ Sb,
                 const float* __restrict__ bnp, const float* __restrict__ bias,
                 const float* __restrict__ resid, float* __restrict__ outF32,
                 _Float16* __restrict__ outSpk, int K, int M, int Mtiles,
                 int doLif) {
    const int lane = threadIdx.x & 31;
    const int blk  = blockIdx.x;
    const int NTP  = NB16 / NW;                 // 8 n-tile pairs
    const int ntp  = blk & (NTP - 1);
    const int mtp  = (blk >> 3) % (Mtiles / MW);
    const int bidx = blk / ((Mtiles / MW) * NTP);
    const int KB   = K >> 5;

    v8f zero = {};
    v8f acc[MW][NW][TT];
#pragma unroll
    for (int mw = 0; mw < MW; ++mw)
#pragma unroll
        for (int nw = 0; nw < NW; ++nw)
#pragma unroll
            for (int t = 0; t < TT; ++t) acc[mw][nw][t] = zero;

    for (int kk = 0; kk < KB; ++kk) {
        // A tiles: one aligned 32B load per lane per m-tile; reused by 8 WMMAs
        v16h a[MW];
#pragma unroll
        for (int mw = 0; mw < MW; ++mw) {
            const _Float16* ap = Wb
                + ((size_t)((mtp * MW + mw) * KB + kk) * 32 + lane) * 16;
            a[mw] = *(const v16h*)ap;
            if (kk + 1 < KB) __builtin_prefetch(ap + 512, 0, 0);
        }
#pragma unroll
        for (int t = 0; t < TT; ++t) {
#pragma unroll
            for (int nw = 0; nw < NW; ++nw) {
                const _Float16* sp = Sb + (size_t)(t * BB + bidx) * K * NN
                    + (size_t)(kk * NB16 + (ntp * NW + nw)) * 512 + lane * 16;
                v16h bm = *(const v16h*)sp;
#pragma unroll
                for (int mw = 0; mw < MW; ++mw)
                    acc[mw][nw][t] = __builtin_amdgcn_wmma_f32_16x16x32_f16(
                        false, a[mw], false, bm, (short)0, acc[mw][nw][t],
                        false, false);
            }
        }
    }

    // ---- epilogue: BN params for this lane's output rows -------------------
    float scale[MW][8], shift[MW][8];
#pragma unroll
    for (int mw = 0; mw < MW; ++mw)
#pragma unroll
        for (int j = 0; j < 8; ++j) {
            int d = (mtp * MW + mw) * 16 + ((lane >> 4) * 8) + j;
            float g  = bnp[d];
            float be = bnp[M + d];
            float mm = bnp[2 * M + d];
            float vv = bnp[3 * M + d];
            float sc = g * rsqrtf(vv + BN_EPS);
            float sh = be - mm * sc;
            if (bias) sh += bias[d] * sc;   // bn(z+bias) = z*sc + (bias-m)*sc + b
            scale[mw][j] = sc;
            shift[mw][j] = sh;
        }

    float vst[MW][NW][8] = {};
#pragma unroll
    for (int t = 0; t < TT; ++t) {
#pragma unroll
        for (int mw = 0; mw < MW; ++mw)
#pragma unroll
            for (int nw = 0; nw < NW; ++nw)
#pragma unroll
                for (int j = 0; j < 8; ++j) {
                    int d  = (mtp * MW + mw) * 16 + ((lane >> 4) * 8) + j;
                    int ng = (ntp * NW + nw) * 16 + (lane & 15);
                    float z = acc[mw][nw][t][j] * scale[mw][j] + shift[mw][j];
                    size_t rmo = (((size_t)t * BB + bidx) * M + d) * NN + ng;
                    if (resid)  z += resid[rmo];
                    if (outF32) outF32[rmo] = z;
                    if (doLif) {
                        float* v = &vst[mw][nw][j];
                        *v = 0.5f * (*v + z);                 // tau = 2
                        float s = (*v >= 1.0f) ? 1.0f : 0.0f;
                        *v *= (1.0f - s);
                        outSpk[(size_t)(t * BB + bidx) * M * NN + sblk(d, ng)]
                            = (_Float16)s;
                    }
                }
    }
}

// ---------------------------------------------------------------------------
// attn path: qsum over dh=48 per head, LIF (vth=0.5) over T, y = attn * k.
// In blocked layout, 16 consecutive channels at fixed n are contiguous and
// each head's 48 channels start 16-aligned -> 3 aligned v16h loads per tensor.
// ---------------------------------------------------------------------------
__global__ void attn_y(const _Float16* __restrict__ q,
                       const _Float16* __restrict__ k,
                       _Float16* __restrict__ y) {
    int tid = blockIdx.x * blockDim.x + threadIdx.x;
    if (tid >= BB * 8 * NN) return;
    int n = tid & (NN - 1);
    int h = (tid >> 8) & 7;
    int b = tid >> 11;
    const int dh = CC / 8;                 // 48 = 3 groups of 16
    float v = 0.f;
#pragma unroll
    for (int t = 0; t < TT; ++t) {
        size_t base = (size_t)(t * BB + b) * CC * NN;
        float qs = 0.f;
#pragma unroll
        for (int g = 0; g < 3; ++g) {
            size_t o = base + sblk(h * dh + g * 16, n);
            v16h qv = *(const v16h*)(q + o);
#pragma unroll
            for (int i = 0; i < 16; ++i) qs += (float)qv[i];
        }
        v = 0.5f * (v + qs);
        float s = (v >= 0.5f) ? 1.0f : 0.0f;   // vth = 0.5
        v *= (1.0f - s);
        v16h zero = {};
#pragma unroll
        for (int g = 0; g < 3; ++g) {
            size_t o = base + sblk(h * dh + g * 16, n);
            v16h kv = *(const v16h*)(k + o);
            *(v16h*)(y + o) = (s != 0.f) ? kv : zero;
        }
    }
}

// ---------------------------------------------------------------------------
extern "C" void kernel_launch(void* const* d_in, const int* in_sizes, int n_in,
                              void* d_out, int out_size, void* d_ws, size_t ws_size,
                              hipStream_t stream) {
    const float* x    = (const float*)d_in[0];
    const float* qw   = (const float*)d_in[1];
    const float* bn_q = (const float*)d_in[2];
    const float* kw   = (const float*)d_in[3];
    const float* bn_k = (const float*)d_in[4];
    const float* pw   = (const float*)d_in[5];
    const float* pb   = (const float*)d_in[6];
    const float* bn_p = (const float*)d_in[7];
    const float* f1w  = (const float*)d_in[8];
    const float* f1b  = (const float*)d_in[9];
    const float* bn1  = (const float*)d_in[10];
    const float* f2w  = (const float*)d_in[11];
    const float* f2b  = (const float*)d_in[12];
    const float* bn2  = (const float*)d_in[13];
    float* out = (float*)d_out;

    // ---- workspace carve-up (all 32B-aligned) -----------------------------
    char* ws = (char*)d_ws;
    _Float16* qw16  = (_Float16*)ws; ws += (size_t)CC * CC * 2;
    _Float16* kw16  = (_Float16*)ws; ws += (size_t)CC * CC * 2;
    _Float16* pw16  = (_Float16*)ws; ws += (size_t)CC * CC * 2;
    _Float16* f1w16 = (_Float16*)ws; ws += (size_t)HID * CC * 2;
    _Float16* f2w16 = (_Float16*)ws; ws += (size_t)CC * HID * 2;
    _Float16* xs16  = (_Float16*)ws; ws += (size_t)TT * BB * CC * NN * 2;
    _Float16* q16   = (_Float16*)ws; ws += (size_t)TT * BB * CC * NN * 2;
    _Float16* k16   = (_Float16*)ws; ws += (size_t)TT * BB * CC * NN * 2;
    _Float16* y16   = (_Float16*)ws; ws += (size_t)TT * BB * CC * NN * 2;
    _Float16* s16   = (_Float16*)ws; ws += (size_t)TT * BB * CC * NN * 2;
    _Float16* s2_16 = (_Float16*)ws; ws += (size_t)TT * BB * HID * NN * 2;
    float*    x2    = (float*)ws;    ws += (size_t)TT * BB * CC * NN * 4;

    // ---- 1) pack weights into WMMA A-operand layout -----------------------
    {
        int n1 = CC * CC, n2 = HID * CC;
        pack_w<<<(n1 + 255) / 256, 256, 0, stream>>>(qw,  qw16,  CC,  CC);
        pack_w<<<(n1 + 255) / 256, 256, 0, stream>>>(kw,  kw16,  CC,  CC);
        pack_w<<<(n1 + 255) / 256, 256, 0, stream>>>(pw,  pw16,  CC,  CC);
        pack_w<<<(n2 + 255) / 256, 256, 0, stream>>>(f1w, f1w16, HID, CC);
        pack_w<<<(n2 + 255) / 256, 256, 0, stream>>>(f2w, f2w16, CC,  HID);
    }

    // ---- 2) LIF on input -> xs spikes -------------------------------------
    {
        int n = BB * CC * NN;
        lif_input<<<(n + 255) / 256, 256, 0, stream>>>(x, xs16);
    }

    const int MT_C = CC  / 16;                      // 24
    const int MT_H = HID / 16;                      // 96
    const int gC   = BB * (MT_C / MW) * (NB16 / NW);   // 16*12*8 = 1536
    const int gH   = BB * (MT_H / MW) * (NB16 / NW);   // 16*48*8 = 6144

    // ---- 3) q and k projections: WMMA + BN + LIF --------------------------
    gemm_bn_lif<<<gC, 32, 0, stream>>>(qw16, xs16, bn_q, nullptr, nullptr,
                                       nullptr, q16, CC, CC, MT_C, 1);
    gemm_bn_lif<<<gC, 32, 0, stream>>>(kw16, xs16, bn_k, nullptr, nullptr,
                                       nullptr, k16, CC, CC, MT_C, 1);

    // ---- 4) head-sum + attn LIF (vth=0.5) + mask k -> y -------------------
    {
        int n = BB * 8 * NN;
        attn_y<<<(n + 255) / 256, 256, 0, stream>>>(q16, k16, y16);
    }

    // ---- 5) proj: WMMA(pw) + pb + BN + residual(x) -> x2 (f32) + LIF -> s --
    gemm_bn_lif<<<gC, 32, 0, stream>>>(pw16, y16, bn_p, pb, x,
                                       x2, s16, CC, CC, MT_C, 1);

    // ---- 6) MLP fc1: WMMA(f1w) + f1b + BN + LIF -> s2 ---------------------
    gemm_bn_lif<<<gH, 32, 0, stream>>>(f1w16, s16, bn1, f1b, nullptr,
                                       nullptr, s2_16, CC, HID, MT_H, 1);

    // ---- 7) MLP fc2: WMMA(f2w) + f2b + BN + residual(x2) -> out -----------
    gemm_bn_lif<<<gC, 32, 0, stream>>>(f2w16, s2_16, bn2, f2b, x2,
                                       out, nullptr, HID, CC, MT_C, 0);
}
